// GATConv_with_22565758173851
// MI455X (gfx1250) — matrix operand order repaired and checked by reading, not compiled, not run on hardware
//
#include <hip/hip_runtime.h>

#define N_NODES   50000
#define N_EDGES   1600000
#define D_IN      256
#define N_HEADS   4
#define D_OUT     32
#define HD        128       // N_HEADS * D_OUT
#define NEG_SLOPE 0.2f

typedef __attribute__((ext_vector_type(16))) __bf16 v16bf;
typedef __attribute__((ext_vector_type(8)))  float  v8f;

__device__ __forceinline__ float leaky(float x) { return x > 0.f ? x : NEG_SLOPE * x; }

// Native-atomic float max: positive floats compare as ints, negative as reversed uints.
// Lowers to global_atomic_max_i32 / global_atomic_min_u32 (single L2 round trip, no CAS loop).
__device__ __forceinline__ void atomic_max_f32(float* addr, float v) {
  if (v >= 0.f) atomicMax((int*)addr, __float_as_int(v));
  else          atomicMin((unsigned int*)addr, __float_as_uint(v));
}

// ---------------------------------------------------------------------------
// 1) h = feat @ fc_w.T  via v_wmma_f32_16x16x32_bf16.
//    One wave per 16x16 output tile. Grid = (50000/16) * (128/16) = 25000 waves.
//    A (16x32 bf16, ISA 7.12.2): lane L: row = L&15, half = L>>4;
//      elems 0..7  -> K = kb + half*8 + e
//      elems 8..15 -> K = kb + 16 + half*8 + (e-8)
//    B (32x16 bf16): lane L: col = L&15, half = L>>4; same K mapping as A
//      (column n of B == row n of fc_w, contiguous in K -> coalesced fp32 loads)
//    C/D (16x16 f32): lane L, vgpr g -> row = half*8 + g, col = L&15
// ---------------------------------------------------------------------------
__global__ void gat_gemm_wmma(const float* __restrict__ feat,
                              const float* __restrict__ fcw,
                              float* __restrict__ hbuf) {
  const int wave = (blockIdx.x * blockDim.x + threadIdx.x) >> 5;
  const int lane = threadIdx.x & 31;
  const int col_tile = wave & 7;    // 128 / 16 = 8 column tiles
  const int row_tile = wave >> 3;
  if (row_tile >= N_NODES / 16) return;   // exact grid: never taken, wave-uniform anyway

  const int r    = lane & 15;
  const int half = lane >> 4;
  const float* __restrict__ arow = feat + (size_t)(row_tile * 16 + r) * D_IN;
  const float* __restrict__ brow = fcw  + (size_t)(col_tile * 16 + r) * D_IN;

  v8f acc = {0.f, 0.f, 0.f, 0.f, 0.f, 0.f, 0.f, 0.f};
#pragma unroll
  for (int kb = 0; kb < D_IN; kb += 32) {
    v16bf a, b;
#pragma unroll
    for (int e = 0; e < 8; ++e) {
      a[e]     = (__bf16)arow[kb + half * 8 + e];
      a[e + 8] = (__bf16)arow[kb + 16 + half * 8 + e];
    }
#pragma unroll
    for (int e = 0; e < 8; ++e) {
      b[e]     = (__bf16)brow[kb + half * 8 + e];
      b[e + 8] = (__bf16)brow[kb + 16 + half * 8 + e];
    }
    acc = __builtin_amdgcn_wmma_f32_16x16x32_bf16(
        /*neg_a=*/false, a, /*neg_b=*/false, b,
        /*c_mod=*/(short)0, acc, /*reuse_a=*/false, /*reuse_b=*/false);
  }

  float* __restrict__ dptr =
      hbuf + (size_t)(row_tile * 16 + half * 8) * HD + col_tile * 16 + r;
#pragma unroll
  for (int g = 0; g < 8; ++g) dptr[(size_t)g * HD] = acc[g];
}

// ---------------------------------------------------------------------------
// 2) el[n,h] = sum_d h[n,h,d]*attn_l[h,d] ;  er likewise. One thread per (n,h).
// ---------------------------------------------------------------------------
__global__ void gat_el_er(const float* __restrict__ hbuf,
                          const float* __restrict__ attn_l,
                          const float* __restrict__ attn_r,
                          float* __restrict__ el, float* __restrict__ er) {
  const int idx = blockIdx.x * blockDim.x + threadIdx.x;  // node*4 + head
  if (idx >= N_NODES * N_HEADS) return;
  const int node = idx >> 2, head = idx & 3;
  const float* __restrict__ hp = hbuf + (size_t)node * HD + head * D_OUT;
  const float* __restrict__ al = attn_l + head * D_OUT;
  const float* __restrict__ ar = attn_r + head * D_OUT;
  float sl = 0.f, sr = 0.f;
#pragma unroll
  for (int d = 0; d < D_OUT; ++d) {
    const float v = hp[d];
    sl += v * al[d];
    sr += v * ar[d];
  }
  el[idx] = sl;
  er[idx] = sr;
}

// ---------------------------------------------------------------------------
// 3) Re-init mutable state every call (harness poisons buffers):
//    out = bias broadcast, m = -3e38, denom = 0.
// ---------------------------------------------------------------------------
__global__ void gat_init(float* __restrict__ out, const float* __restrict__ bias,
                         float* __restrict__ m, float* __restrict__ denom) {
  const int idx = blockIdx.x * blockDim.x + threadIdx.x;
  if (idx < N_NODES * HD) out[idx] = bias[idx & (HD - 1)];
  if (idx < N_NODES * N_HEADS) {
    m[idx] = -3.0e38f;
    denom[idx] = 0.f;
  }
}

// ---------------------------------------------------------------------------
// 4) Segment max of leaky(el[src]+er[dst]) over dst. One thread per edge.
// ---------------------------------------------------------------------------
__global__ void gat_edge_max(const int* __restrict__ src, const int* __restrict__ dst,
                             const float* __restrict__ el, const float* __restrict__ er,
                             float* __restrict__ m) {
  const int e = blockIdx.x * blockDim.x + threadIdx.x;
  if (e >= N_EDGES) return;
  const int s = src[e], d = dst[e];
#pragma unroll
  for (int hh = 0; hh < N_HEADS; ++hh) {
    const float v = leaky(el[s * N_HEADS + hh] + er[d * N_HEADS + hh]);
    atomic_max_f32(m + d * N_HEADS + hh, v);
  }
}

// ---------------------------------------------------------------------------
// 5) denom[dst] += exp(e - m[dst]).  Recomputes e (L2-resident gathers) instead
//    of spilling a 25.6MB per-edge buffer.
// ---------------------------------------------------------------------------
__global__ void gat_edge_expsum(const int* __restrict__ src, const int* __restrict__ dst,
                                const float* __restrict__ el, const float* __restrict__ er,
                                const float* __restrict__ m, float* __restrict__ denom) {
  const int e = blockIdx.x * blockDim.x + threadIdx.x;
  if (e >= N_EDGES) return;
  const int s = src[e], d = dst[e];
#pragma unroll
  for (int hh = 0; hh < N_HEADS; ++hh) {
    const float v  = leaky(el[s * N_HEADS + hh] + er[d * N_HEADS + hh]);
    const float ex = __expf(v - m[d * N_HEADS + hh]);
    atomicAdd(denom + d * N_HEADS + hh, ex);
  }
}

// ---------------------------------------------------------------------------
// 6) out[dst] += h[src] * a.  One wave per edge; lane l covers features
//    {l, l+32, l+64, l+96} (head = chunk index). 4 global_atomic_add_f32/lane.
// ---------------------------------------------------------------------------
__global__ void gat_edge_aggr(const int* __restrict__ src, const int* __restrict__ dst,
                              const float* __restrict__ el, const float* __restrict__ er,
                              const float* __restrict__ m, const float* __restrict__ denom,
                              const float* __restrict__ hbuf, float* __restrict__ out) {
  const int e = blockIdx.x * 8 + (threadIdx.x >> 5);
  const int lane = threadIdx.x & 31;
  if (e >= N_EDGES) return;
  const int s = src[e], d = dst[e];
  const float* __restrict__ hs = hbuf + (size_t)s * HD;
  float* __restrict__ od = out + (size_t)d * HD;

  float a[N_HEADS];
#pragma unroll
  for (int hh = 0; hh < N_HEADS; ++hh) {
    const float v  = leaky(el[s * N_HEADS + hh] + er[d * N_HEADS + hh]);
    const float ex = __expf(v - m[d * N_HEADS + hh]);
    a[hh] = ex / fmaxf(denom[d * N_HEADS + hh], 1e-9f);
  }
#pragma unroll
  for (int i = 0; i < N_HEADS; ++i) {
    const int j = i * 32 + lane;
    atomicAdd(od + j, hs[j] * a[i]);
  }
}

// ---------------------------------------------------------------------------
extern "C" void kernel_launch(void* const* d_in, const int* in_sizes, int n_in,
                              void* d_out, int out_size, void* d_ws, size_t ws_size,
                              hipStream_t stream) {
  const float* feat   = (const float*)d_in[0];
  const float* fcw    = (const float*)d_in[1];
  const float* attn_l = (const float*)d_in[2];
  const float* attn_r = (const float*)d_in[3];
  const float* bias   = (const float*)d_in[4];
  const int*   src    = (const int*)d_in[5];
  const int*   dst    = (const int*)d_in[6];
  float* out = (float*)d_out;

  // Workspace layout (28.8 MB total):
  float* ws    = (float*)d_ws;
  float* hbuf  = ws;                          // N * 128
  float* el    = hbuf + (size_t)N_NODES * HD; // N * 4
  float* er    = el + N_NODES * N_HEADS;      // N * 4
  float* m     = er + N_NODES * N_HEADS;      // N * 4
  float* denom = m  + N_NODES * N_HEADS;      // N * 4

  const int BLK = 256;

  // 1) GEMM: 25000 waves = 3125 blocks of 8 waves.
  gat_gemm_wmma<<<(N_NODES / 16) * 8 * 32 / BLK, BLK, 0, stream>>>(feat, fcw, hbuf);

  // 2) Attention logits.
  gat_el_er<<<(N_NODES * N_HEADS + BLK - 1) / BLK, BLK, 0, stream>>>(hbuf, attn_l, attn_r, el, er);

  // 3) Init out/m/denom (deterministic per call).
  gat_init<<<(N_NODES * HD + BLK - 1) / BLK, BLK, 0, stream>>>(out, bias, m, denom);

  // 4) Segment max.
  gat_edge_max<<<(N_EDGES + BLK - 1) / BLK, BLK, 0, stream>>>(src, dst, el, er, m);

  // 5) Segment sum of exp.
  gat_edge_expsum<<<(N_EDGES + BLK - 1) / BLK, BLK, 0, stream>>>(src, dst, el, er, m, denom);

  // 6) Weighted scatter-aggregate (wave per edge).
  gat_edge_aggr<<<N_EDGES / 8, BLK, 0, stream>>>(src, dst, el, er, m, denom, hbuf, out);
}
